// ConvSE3_53523882442961
// MI455X (gfx1250) — matrix operand order repaired and checked by reading, not compiled, not run on hardware
//
#include <hip/hip_runtime.h>
#include <math.h>

// ---------------------------------------------------------------------------
// SE(3)-equivariant conv (ConvSE3) for MI455X / gfx1250, wave32 + WMMA bf16.
//
// Shapes: B=2, N=512, K=32, M=16, MID=128, degrees (0,1).
// Dominant compute: per-edge radial MLPs -> batched GEMMs
//   [32768,128] x [128,128]  and  [32768,128] x [128, nf*256]
// mapped onto v_wmma_f32_16x16x32_bf16 (f32 accumulate).
//
// L3 uses 32-edge tiles with dual accumulators per wave (each B fragment
// feeds 2 WMMAs -> halves the dominant L2 B-stream), A tile staged into LDS
// via global_load_async_to_lds_b128 (ASYNCcnt).
// ---------------------------------------------------------------------------

typedef __attribute__((ext_vector_type(16))) __bf16 v16bf;
typedef __attribute__((ext_vector_type(8)))  float  v8f;

static_assert(sizeof(__bf16) == 2, "bf16 must be 2 bytes");

#define BB   2
#define NN   512
#define KK   32
#define MM   16
#define MIDC 128
#define NEDGE (BB * NN * KK)          // 32768

__device__ __forceinline__ float gelu_f(float x) {
  // exact gelu (approximate=False): 0.5*x*(1+erf(x/sqrt(2)))
  return 0.5f * x * (1.0f + erff(x * 0.70710678118654752440f));
}

// A fragment (16-bit 16x32, ISA 7.12.2): lane half h, runs k0+8h..+7 and k0+16+8h..+7
__device__ __forceinline__ v16bf load_a_frag(const __bf16* p /* row + k0 + 8*half */) {
  v16bf v;
#pragma unroll
  for (int j = 0; j < 8; ++j) v[j] = p[j];
#pragma unroll
  for (int j = 0; j < 8; ++j) v[8 + j] = p[16 + j];
  return v;
}

// B fragment (32x16): lane half h, runs k0+8h..+7 and k0+16+8h..+7
__device__ __forceinline__ v16bf load_b_frag(const __bf16* p /* col + k0 + 8*half */) {
  v16bf v;
#pragma unroll
  for (int j = 0; j < 8; ++j) v[j] = p[j];
#pragma unroll
  for (int j = 0; j < 8; ++j) v[8 + j] = p[16 + j];
  return v;
}

// Async 16-byte global -> LDS copy (per active lane), tracked by ASYNCcnt.
__device__ __forceinline__ void async_copy16(unsigned lds_off, const void* gptr) {
  asm volatile("global_load_async_to_lds_b128 %0, %1, off"
               :
               : "v"(lds_off), "v"((unsigned long long)(size_t)gptr)
               : "memory");
}

// ---------------------------------------------------------------------------
// Prep: zero accumulators; transpose + convert w2/w3 to bf16 [n][k] layout.
// ---------------------------------------------------------------------------
struct PrepArgs {
  const float* w2[4];
  const float* w3[4];
  __bf16* w2T;            // [4][128*128], pair p at p*16384, element [n*128+k]
  __bf16* w3T[4];         // pair p: [NCp*128], element [n*128+k]
  float* acc0;            // [NEDGE*16]
  float* acc1;            // [NEDGE*48]
};

__global__ void prep_kernel(PrepArgs a) {
  const size_t tid = (size_t)blockIdx.x * blockDim.x + threadIdx.x;
  const size_t stride = (size_t)gridDim.x * blockDim.x;
  for (size_t i = tid; i < (size_t)NEDGE * 16; i += stride) a.acc0[i] = 0.0f;
  for (size_t i = tid; i < (size_t)NEDGE * 48; i += stride) a.acc1[i] = 0.0f;
#pragma unroll
  for (int p = 0; p < 4; ++p) {
    for (size_t i = tid; i < 16384; i += stride) {
      const int n = (int)(i >> 7), k = (int)(i & 127);
      a.w2T[p * 16384 + i] = (__bf16)a.w2[p][(size_t)k * 128 + n];
    }
  }
#pragma unroll
  for (int p = 0; p < 4; ++p) {
    const int NC = (p == 3) ? 768 : 256;
    for (size_t i = tid; i < (size_t)NC * 128; i += stride) {
      const int n = (int)(i >> 7), k = (int)(i & 127);
      a.w3T[p][i] = (__bf16)a.w3[p][(size_t)k * NC + n];
    }
  }
}

// ---------------------------------------------------------------------------
// Layer 1: h1 = gelu(LN(dist * w1 + b1))  -> bf16 [NEDGE,128]
// ---------------------------------------------------------------------------
__global__ void l1_kernel(const float* __restrict__ dist,
                          const float* __restrict__ w1,
                          const float* __restrict__ b1,
                          const float* __restrict__ g1,
                          const float* __restrict__ be1,
                          __bf16* __restrict__ h1) {
  const int e = blockIdx.x * blockDim.x + threadIdx.x;  // grid covers exactly NEDGE
  const float d = dist[e];
  float s1 = 0.0f, s2 = 0.0f;
#pragma unroll 8
  for (int c = 0; c < MIDC; ++c) {
    const float t = fmaf(d, w1[c], b1[c]);
    s1 += t; s2 += t * t;
  }
  const float mu = s1 * (1.0f / MIDC);
  const float var = s2 * (1.0f / MIDC) - mu * mu;
  const float rs = rsqrtf(var + 1e-5f);
  __bf16* row = h1 + (size_t)e * MIDC;
#pragma unroll 8
  for (int c = 0; c < MIDC; ++c) {
    const float t = fmaf(d, w1[c], b1[c]);
    row[c] = (__bf16)gelu_f((t - mu) * rs * g1[c] + be1[c]);
  }
}

// ---------------------------------------------------------------------------
// Layer 2: h2 = gelu(LN(h1 @ w2 + b2)) -> bf16 [NEDGE,128]
// One block = 16 edges x 128 cols. 4 waves, 2 col-tiles each, 4 WMMA k-steps.
// ---------------------------------------------------------------------------
__global__ void __launch_bounds__(128)
l2_kernel(const __bf16* __restrict__ h1,
          const __bf16* __restrict__ w2T,
          const float* __restrict__ b2,
          const float* __restrict__ g2,
          const float* __restrict__ be2,
          __bf16* __restrict__ h2) {
  __shared__ float Cl[16][128];
  __shared__ float red[16][8][2];

  const int tid = threadIdx.x;
  const int lane = tid & 31;
  const int wave = tid >> 5;
  const int r = lane & 15;
  const int half = lane >> 4;
  const int e0 = blockIdx.x * 16;

  __builtin_prefetch(w2T, 0, 0);  // global_prefetch_b8

  v16bf afrag[4];
  const __bf16* arow = h1 + (size_t)(e0 + r) * MIDC + 8 * half;
#pragma unroll
  for (int kk = 0; kk < 4; ++kk) afrag[kk] = load_a_frag(arow + 32 * kk);

#pragma unroll
  for (int t = 0; t < 2; ++t) {
    const int ct = wave * 2 + t;                 // col tile 0..7
    v8f c = {};
    const __bf16* bcol = w2T + (size_t)(ct * 16 + r) * MIDC + 8 * half;
#pragma unroll
    for (int kk = 0; kk < 4; ++kk) {
      const v16bf bfrag = load_b_frag(bcol + 32 * kk);
      c = __builtin_amdgcn_wmma_f32_16x16x32_bf16(false, afrag[kk], false, bfrag,
                                                  (short)0, c, false, false);
    }
    const float bias = b2[ct * 16 + r];
#pragma unroll
    for (int v = 0; v < 8; ++v)
      Cl[half * 8 + v][ct * 16 + r] = c[v] + bias;   // D row = v + 8*half, col = lane&15
  }
  __syncthreads();

  // LayerNorm over 128 cols per row, then gelu, convert to bf16.
  const int row = tid >> 3;   // 0..15
  const int sub = tid & 7;    // 0..7 -> 16 cols each
  float s1 = 0.0f, s2 = 0.0f;
#pragma unroll
  for (int j = 0; j < 16; ++j) {
    const float v = Cl[row][sub * 16 + j];
    s1 += v; s2 += v * v;
  }
  red[row][sub][0] = s1; red[row][sub][1] = s2;
  __syncthreads();
  float ts1 = 0.0f, ts2 = 0.0f;
#pragma unroll
  for (int j = 0; j < 8; ++j) { ts1 += red[row][j][0]; ts2 += red[row][j][1]; }
  const float mu = ts1 * (1.0f / MIDC);
  const float var = ts2 * (1.0f / MIDC) - mu * mu;
  const float rs = rsqrtf(var + 1e-5f);
  __bf16* orow = h2 + (size_t)(e0 + row) * MIDC;
#pragma unroll
  for (int j = 0; j < 16; ++j) {
    const int cidx = sub * 16 + j;
    const float v = (Cl[row][cidx] - mu) * rs * g2[cidx] + be2[cidx];
    orow[cidx] = (__bf16)gelu_f(v);
  }
}

// ---------------------------------------------------------------------------
// Layer 3 fused: R = h2 @ w3 + b3 (WMMA -> LDS, never to HBM), then
//   y[i,od,f]  = sum_oi basis[od,oi,f] * xg[i,oi]        (gathered neighbor feats)
//   acc[o,od] += sum_{i,f} R[o*16*NF + i*NF + f] * y[i,od,f]
// One block = ET(=32) edges: 2 row-tiles share every B fragment (2 WMMAs per
// B load -> halves L2 B traffic). A tile staged via async global->LDS copies.
// Dynamic LDS (up to ~125 KB for pair (1,1); WGP has 320 KB).
// ---------------------------------------------------------------------------
template <int NF, int ODO, int ODI, int ET>
__global__ void __launch_bounds__(128)
l3_kernel(const __bf16* __restrict__ h2,
          const __bf16* __restrict__ w3T,
          const float* __restrict__ b3,
          const float* __restrict__ basis,
          const float* __restrict__ xin,
          const int* __restrict__ nbr,
          float* __restrict__ acc) {
  constexpr int NC  = MM * MM * NF;      // 256 or 768 output cols of w3
  constexpr int NT  = NC / 16;           // 16 or 48 col tiles
  constexpr int NY  = MM * ODO * NF;     // y size per edge
  constexpr int NO  = MM * ODO;          // outputs per edge
  constexpr int NRT = ET / 16;           // row tiles per block
  constexpr int TPE = 128 / ET;          // threads per edge (phase 2)
  static_assert(NT % 4 == 0 && 128 % ET == 0, "tiling");

  extern __shared__ char smem3[];
  float*  Rlds = (float*)smem3;                 // [ET][NC]
  float*  Ylds = Rlds + ET * NC;                // [ET][NY]
  __bf16* Alds = (__bf16*)(Ylds + ET * NY);     // [ET][MIDC]

  const int tid = threadIdx.x;
  const int lane = tid & 31;
  const int wave = tid >> 5;
  const int r = lane & 15;
  const int half = lane >> 4;
  const int e0 = blockIdx.x * ET;

  __builtin_prefetch(w3T, 0, 0);  // global_prefetch_b8

  // --- Stage A tile (ET x 128 bf16) into LDS via async copies -------------
  {
    const char* gsrc = (const char*)(h2 + (size_t)e0 * MIDC);
    const unsigned abase = (unsigned)(size_t)Alds;
    constexpr int NCHUNK = ET * MIDC * 2 / 16;   // 16B chunks
    for (int c = tid; c < NCHUNK; c += 128)
      async_copy16(abase + c * 16, gsrc + (size_t)c * 16);
    asm volatile("s_wait_asynccnt 0x0" ::: "memory");
  }
  __syncthreads();

  // --- Phase 1: WMMA  R = h2_tile @ w3T (+b3)  into LDS -------------------
  v16bf afrag[NRT][4];
#pragma unroll
  for (int rt = 0; rt < NRT; ++rt)
#pragma unroll
    for (int kk = 0; kk < 4; ++kk)
      afrag[rt][kk] = load_a_frag(Alds + (rt * 16 + r) * MIDC + kk * 32 + 8 * half);

  for (int ct = wave; ct < NT; ct += 4) {      // wave-uniform loop, NT % 4 == 0
    v8f c[NRT] = {};
    const __bf16* bcol = w3T + (size_t)(ct * 16 + r) * MIDC + 8 * half;
#pragma unroll
    for (int kk = 0; kk < 4; ++kk) {
      const v16bf bfrag = load_b_frag(bcol + 32 * kk);
#pragma unroll
      for (int rt = 0; rt < NRT; ++rt)
        c[rt] = __builtin_amdgcn_wmma_f32_16x16x32_bf16(false, afrag[rt][kk], false,
                                                        bfrag, (short)0, c[rt],
                                                        false, false);
    }
    const float bias = b3[ct * 16 + r];
#pragma unroll
    for (int rt = 0; rt < NRT; ++rt)
#pragma unroll
      for (int v = 0; v < 8; ++v)
        Rlds[(rt * 16 + half * 8 + v) * NC + ct * 16 + r] = c[rt][v] + bias;
  }
  __syncthreads();

  // --- Phase 2: basis contraction -----------------------------------------
  const int le = tid / TPE;     // local edge 0..ET-1
  const int sub = tid % TPE;
  const int e = e0 + le;
  const int b = e >> 14;                         // e / (N*K)
  const int j = nbr[e];                          // neighbor node index
  const float* bas = basis + (size_t)e * (ODO * ODI * NF);
  const float* xg = xin + ((size_t)(b * NN + j) * MM) * ODI;

  for (int yi = sub; yi < NY; yi += TPE) {
    const int i = yi / (ODO * NF);
    const int rem = yi - i * (ODO * NF);
    const int od = rem / NF;
    const int f = rem - od * NF;
    float s = 0.0f;
#pragma unroll
    for (int oi = 0; oi < ODI; ++oi)
      s += bas[(od * ODI + oi) * NF + f] * xg[i * ODI + oi];
    Ylds[le * NY + yi] = s;
  }
  __syncthreads();

  for (int o_ = sub; o_ < NO; o_ += TPE) {
    const int o = o_ / ODO;
    const int od = o_ - o * ODO;
    float s = 0.0f;
#pragma unroll
    for (int i = 0; i < MM; ++i) {
#pragma unroll
      for (int f = 0; f < NF; ++f)
        s += Rlds[le * NC + o * (MM * NF) + i * NF + f] *
             Ylds[le * NY + i * (ODO * NF) + od * NF + f];
    }
    acc[(size_t)e * NO + o_] += s;   // pairs with same 'do' are stream-ordered
  }
}

// ---------------------------------------------------------------------------
// Pool: masked mean over K neighbors + self-interaction, write outputs.
// out = [out0: B*N*16] ++ [out1: B*N*48]
// ---------------------------------------------------------------------------
__global__ void __launch_bounds__(64)
pool_kernel(const float* __restrict__ acc0, const float* __restrict__ acc1,
            const float* __restrict__ x0, const float* __restrict__ x1,
            const float* __restrict__ si0, const float* __restrict__ si1,
            const unsigned char* __restrict__ mask,
            float* __restrict__ out) {
  const int node = blockIdx.x;          // 0 .. B*N-1
  const int t = threadIdx.x;            // 0 .. 63
  const unsigned char* mrow = mask + (size_t)node * KK;
  float den = 0.0f;
#pragma unroll 8
  for (int k = 0; k < KK; ++k) den += mrow[k] ? 1.0f : 0.0f;
  den = fmaxf(den, 1.0f);

  if (t < 16) {
    const int m = t;
    float s = 0.0f;
    for (int k = 0; k < KK; ++k)
      if (mrow[k]) s += acc0[((size_t)node * KK + k) * 16 + m];
    float si = 0.0f;
#pragma unroll
    for (int d = 0; d < MM; ++d) si += x0[node * MM + d] * si0[d * MM + m];
    out[node * 16 + m] = s / den + si;
  } else {
    const int q = t - 16;                // 0..47
    const int m = q / 3, od = q % 3;
    float s = 0.0f;
    for (int k = 0; k < KK; ++k)
      if (mrow[k]) s += acc1[((size_t)node * KK + k) * 48 + m * 3 + od];
    float si = 0.0f;
#pragma unroll
    for (int d = 0; d < MM; ++d) si += x1[(node * MM + d) * 3 + od] * si1[d * MM + m];
    out[BB * NN * 16 + node * 48 + m * 3 + od] = s / den + si;
  }
}

// ---------------------------------------------------------------------------
// Host launcher.
// Input order (dict insertion order, params flattened in insertion order):
//  0 x0, 1 x1, 2 rel_dist, 3 basis_00, 4 basis_01, 5 basis_10, 6 basis_11,
//  7+10p .. rp_p: {w1,b1,g1,be1,w2,b2,g2,be2,w3,b3} for p = (di*2+do) in
//  order (0,0),(0,1),(1,0),(1,1); 47 si_0, 48 si_1, 49 nbr_idx, 50 nbr_mask.
// ---------------------------------------------------------------------------
extern "C" void kernel_launch(void* const* d_in, const int* in_sizes, int n_in,
                              void* d_out, int out_size, void* d_ws, size_t ws_size,
                              hipStream_t stream) {
  (void)in_sizes; (void)n_in; (void)out_size; (void)ws_size;

  const float* x0   = (const float*)d_in[0];
  const float* x1   = (const float*)d_in[1];
  const float* dist = (const float*)d_in[2];
  const float* si0  = (const float*)d_in[47];
  const float* si1  = (const float*)d_in[48];
  const int* nbr    = (const int*)d_in[49];
  const unsigned char* mask = (const unsigned char*)d_in[50];
  auto RP = [&](int p, int which) -> const float* {
    return (const float*)d_in[7 + 10 * p + which];
  };

  // Workspace layout (bytes from base):
  char* ws = (char*)d_ws;
  __bf16* w2T = (__bf16*)(ws);                        // 4*16384 bf16  = 128 KiB
  __bf16* w3T_base = (__bf16*)(ws + 131072);          // 196608 bf16   = 384 KiB
  __bf16* w3T[4] = { w3T_base, w3T_base + 32768, w3T_base + 65536, w3T_base + 98304 };
  __bf16* h1 = (__bf16*)(ws + 524288);                // NEDGE*128 bf16 = 8 MiB
  __bf16* h2 = (__bf16*)(ws + 524288 + 8388608);      // 8 MiB
  float* acc0 = (float*)(ws + 524288 + 2 * 8388608);  // NEDGE*16 f32 = 2 MiB
  float* acc1 = (float*)(ws + 524288 + 2 * 8388608 + 2097152); // NEDGE*48 f32 = 6 MiB

  PrepArgs pa;
  for (int p = 0; p < 4; ++p) {
    pa.w2[p] = RP(p, 4);
    pa.w3[p] = RP(p, 8);
    pa.w3T[p] = w3T[p];
  }
  pa.w2T = w2T; pa.acc0 = acc0; pa.acc1 = acc1;
  prep_kernel<<<512, 256, 0, stream>>>(pa);

  const int nBlkEdge = NEDGE / 16;  // 2048 (layer 2)
  const int nBlk3 = NEDGE / 32;     // 1024 (layer 3, ET=32)
  auto smem3 = [](int NC_, int NY_) -> size_t {
    return (size_t)32 * NC_ * 4 + (size_t)32 * NY_ * 4 + (size_t)32 * MIDC * 2;
  };

  for (int p = 0; p < 4; ++p) {
    l1_kernel<<<NEDGE / 256, 256, 0, stream>>>(dist, RP(p, 0), RP(p, 1), RP(p, 2),
                                               RP(p, 3), h1);
    l2_kernel<<<nBlkEdge, 128, 0, stream>>>(h1, w2T + p * 16384, RP(p, 5), RP(p, 6),
                                            RP(p, 7), h2);
    const float* bas = (const float*)d_in[3 + p];
    const float* xin = (p >> 1) ? x1 : x0;   // di = p>>1
    if (p == 0)
      l3_kernel<1, 1, 1, 32><<<nBlk3, 128, smem3(256, 16), stream>>>(
          h2, w3T[0], RP(0, 9), bas, xin, nbr, acc0);
    else if (p == 1)
      l3_kernel<1, 3, 1, 32><<<nBlk3, 128, smem3(256, 48), stream>>>(
          h2, w3T[1], RP(1, 9), bas, xin, nbr, acc1);
    else if (p == 2)
      l3_kernel<1, 1, 3, 32><<<nBlk3, 128, smem3(256, 48), stream>>>(
          h2, w3T[2], RP(2, 9), bas, xin, nbr, acc0);
    else
      l3_kernel<3, 3, 3, 32><<<nBlk3, 128, smem3(768, 144), stream>>>(
          h2, w3T[3], RP(3, 9), bas, xin, nbr, acc1);
  }

  pool_kernel<<<BB * NN, 64, 0, stream>>>(acc0, acc1, x0, x1, si0, si1, mask,
                                          (float*)d_out);
}
